// SAGE_ks_31997506355387
// MI455X (gfx1250) — matrix operand order repaired and checked by reading, not compiled, not run on hardware
//
#include <hip/hip_runtime.h>
#include <hip/hip_bf16.h>

// ---------------------------------------------------------------------------
// GraphSAGE 3-layer forward for MI455X (gfx1250, wave32).
//   layer: out = relu?( mean_aggr(h[src]->dst) @ Wl + h @ Wr + b )
// Dense transforms: v_wmma_f32_16x16x32_bf16 (f32 accumulate), deg_inv folded
// into the A operand. Weights are pre-packed once to bf16/transposed/K-padded
// so GEMM blocks stage them into LDS with async global->LDS copies
// (ASYNCcnt-tracked), then run the whole K loop barrier-free.
// Aggregation (~3.4 GB traffic -> ~150us @ 23.3 TB/s) dominates runtime; the
// ~7 GFLOP of GEMM is effectively free on the WMMA pipes.
// ---------------------------------------------------------------------------

typedef __attribute__((ext_vector_type(16))) __bf16 v16bf;
typedef __attribute__((ext_vector_type(8)))  float  v8f;

#define N_NODES 100000
#define N_EDGES 1600000
#define DIM_IN  15
#define DIM_H   128

// Toggle: 1 = stage LDS via global_load_async_to_lds_b128 inline asm,
//         0 = plain vector copies (fallback if assembler rejects mnemonic).
#ifndef SAGE_ASYNC_LDS
#define SAGE_ASYNC_LDS 1
#endif

__device__ __forceinline__ __bf16 f2bf(float f) {
    union { float f; unsigned u; } x; x.f = f;
    unsigned u = x.u;
    unsigned r = (u + 0x7FFFu + ((u >> 16) & 1u)) >> 16;   // round-nearest-even
    union { unsigned short s; __bf16 b; } y; y.s = (unsigned short)r;
    return y.b;
}

// ---------------------------------------------------------------------------
__global__ __launch_bounds__(256)
void fill_kernel(float* __restrict__ p, long n, float v) {
    long i = (long)blockIdx.x * 256 + threadIdx.x;
    if (i < n) p[i] = v;
}

__global__ __launch_bounds__(256)
void deg_count_kernel(const int* __restrict__ edge, int nE, float* __restrict__ deg) {
    int e = blockIdx.x * 256 + threadIdx.x;
    if (e < nE) atomicAdd(&deg[edge[nE + e]], 1.0f);     // dst row of edge_index
}

__global__ __launch_bounds__(256)
void deg_inv_kernel(float* __restrict__ deg, int nN) {
    int i = blockIdx.x * 256 + threadIdx.x;
    if (i < nN) { float d = deg[i]; deg[i] = 1.0f / (d > 1.0f ? d : 1.0f); }
}

// Pack Wl|Wr ([K_IN x 128] f32, row-major) into WT[mat][n][kpad] bf16,
// zero-padded past K_IN, so GEMM B-fragments are contiguous aligned 32B runs.
template <int K_IN>
__global__ __launch_bounds__(256)
void pack_weights_kernel(const float* __restrict__ Wl, const float* __restrict__ Wr,
                         __bf16* __restrict__ WT) {
    constexpr int KPAD = (K_IN + 31) & ~31;
    const int total = 2 * DIM_H * KPAD;
    int i = blockIdx.x * 256 + threadIdx.x;
    if (i >= total) return;
    int mat = i / (DIM_H * KPAD);
    int rem = i % (DIM_H * KPAD);
    int n = rem / KPAD, k = rem % KPAD;
    const float* W = mat ? Wr : Wl;
    WT[i] = (k < K_IN) ? f2bf(W[k * DIM_H + n]) : f2bf(0.0f);
}

// scatter-add x[src, 0:15] into agg[dst, 0:15] (agg row stride = 128)
__global__ __launch_bounds__(256)
void scatter15_kernel(const int* __restrict__ edge, int nE,
                      const float* __restrict__ x, float* __restrict__ agg) {
    int e = blockIdx.x * 256 + threadIdx.x;
    if (e >= nE) return;
    long src = edge[e], dst = edge[nE + e];
    const float* xs = x + src * DIM_IN;
    float* ad = agg + dst * DIM_H;
    #pragma unroll
    for (int d = 0; d < DIM_IN; ++d) atomicAdd(&ad[d], xs[d]);
}

// scatter-add h[src, 0:128] into agg[dst, 0:128]; one wave per edge,
// each lane gathers a float4 and issues 4 float atomics.
__global__ __launch_bounds__(256)
void scatter128_kernel(const int* __restrict__ edge, int nE,
                       const float* __restrict__ h, float* __restrict__ agg) {
    long gid  = (long)blockIdx.x * 256 + threadIdx.x;
    int  e    = (int)(gid >> 5);
    int  lane = (int)(gid & 31);
    if (e >= nE) return;
    long src = edge[e], dst = edge[nE + e];
    const float4 v = *(const float4*)&h[src * DIM_H + lane * 4];
    float* ad = agg + dst * DIM_H + lane * 4;
    atomicAdd(&ad[0], v.x);
    atomicAdd(&ad[1], v.y);
    atomicAdd(&ad[2], v.z);
    atomicAdd(&ad[3], v.w);
}

// ---------------------------------------------------------------------------
// Dual-GEMM: Out[N x 128] = relu?( (A1 * deg_inv) @ Wl + A2 @ Wr + bias )
// WT = pre-packed bf16 weights [2][128][KPAD]. One wave owns a 16x128 output
// strip (8 v8f accumulators); the full weight block is staged to LDS once via
// async global->LDS copies, then the K loop runs barrier-free.
// ---------------------------------------------------------------------------
template <int K_IN, bool RELU>
__global__ __launch_bounds__(256)
void sage_gemm_kernel(const float* __restrict__ A1, int lda1,
                      const float* __restrict__ A2, int lda2,
                      const __bf16* __restrict__ WT,
                      const float* __restrict__ bias,
                      const float* __restrict__ deg_inv,
                      float* __restrict__ Out, int nNodes) {
    constexpr int KPAD   = (K_IN + 31) & ~31;
    constexpr int KSTEPS = KPAD / 32;

    // Full weight block: [mat][n][k] bf16 (16 KB for K=32, 64 KB for K=128;
    // CDNA5 LDS is 320 KB per WGP so this fits comfortably).
    __shared__ __align__(32) __bf16 sW[2][DIM_H][KPAD];

    const int tid  = threadIdx.x;
    const int wave = tid >> 5;
    const int lane = tid & 31;
    const int half = lane >> 4;            // 0: lanes 0-15, 1: lanes 16-31
    const int mrow = lane & 15;

    // ---- Stage weights into LDS (raw bf16 copy, 16B chunks) ----------------
    {
        constexpr int CHUNKS = (2 * DIM_H * KPAD * 2) / 16;   // 16B chunks
#if SAGE_ASYNC_LDS
        for (int c = tid; c < CHUNKS; c += 256) {
            unsigned lds_addr = (unsigned)(uintptr_t)
                (__attribute__((address_space(3))) char*)((char*)&sW[0][0][0] + c * 16);
            // GVS mode: LDS-dest VGPR, 32-bit VGPR offset, 64-bit SGPR base.
            asm volatile("global_load_async_to_lds_b128 %0, %1, %2"
                         :: "v"(lds_addr), "v"((unsigned)(c * 16)), "s"(WT)
                         : "memory");
        }
        asm volatile("s_wait_asynccnt 0" ::: "memory");
#else
        const uint4* gsrc = (const uint4*)WT;
        uint4* ldst = (uint4*)&sW[0][0][0];
        for (int c = tid; c < CHUNKS; c += 256) ldst[c] = gsrc[c];
#endif
    }
    __syncthreads();

    // ---- Per-wave 16x128 output strip --------------------------------------
    const int nTiles = nNodes >> 4;        // 100000/16 = 6250 (exact)
    const int tile   = blockIdx.x * 8 + wave;
    if (tile >= nTiles) return;            // wave-uniform; no barriers below
    const int row = tile * 16 + mrow;      // A-fragment row for this lane
    const float scale = deg_inv[row];

    v8f acc[8];
    const v8f vzero = {0, 0, 0, 0, 0, 0, 0, 0};
    #pragma unroll
    for (int t = 0; t < 8; ++t) acc[t] = vzero;

    for (int ks = 0; ks < KSTEPS; ++ks) {
        if (ks + 1 < KSTEPS) {             // -> global_prefetch_b8
            __builtin_prefetch(&A1[(long)row * lda1 + (ks + 1) * 32], 0, 0);
            __builtin_prefetch(&A2[(long)row * lda2 + (ks + 1) * 32], 0, 0);
        }

        // A fragments (16x32 bf16): lane holds row mrow; elements 0..7 -> K =
        // kb..kb+7, elements 8..15 -> K = kb+8..kb+15, kb = ks*32 + half*16.
        v16bf a1, a2;
        const int kb = ks * 32 + half * 16;
        #pragma unroll
        for (int j = 0; j < 8; ++j) {
            int k0 = kb + j, k1 = kb + 8 + j;
            float f1a = (k0 < K_IN) ? A1[(long)row * lda1 + k0] * scale : 0.0f;
            float f1b = (k1 < K_IN) ? A1[(long)row * lda1 + k1] * scale : 0.0f;
            float f2a = (k0 < K_IN) ? A2[(long)row * lda2 + k0] : 0.0f;
            float f2b = (k1 < K_IN) ? A2[(long)row * lda2 + k1] : 0.0f;
            a1[j] = f2bf(f1a); a1[8 + j] = f2bf(f1b);
            a2[j] = f2bf(f2a); a2[8 + j] = f2bf(f2b);
        }

        // B fragments (32x16 bf16): lane = column t*16+mrow, K = half*16 + e;
        // one aligned 32B contiguous LDS read per fragment.
        #pragma unroll
        for (int t = 0; t < 8; ++t) {
            const int col = t * 16 + mrow;
            const v16bf bl = *(const v16bf*)&sW[0][col][ks * 32 + half * 16];
            const v16bf br = *(const v16bf*)&sW[1][col][ks * 32 + half * 16];
            acc[t] = __builtin_amdgcn_wmma_f32_16x16x32_bf16(
                false, a1, false, bl, (short)0, acc[t], false, false);
            acc[t] = __builtin_amdgcn_wmma_f32_16x16x32_bf16(
                false, a2, false, br, (short)0, acc[t], false, false);
        }
    }

    // C/D layout: lane col = lane&15, VGPR r -> row r + 8*half.
    #pragma unroll
    for (int t = 0; t < 8; ++t) {
        const int col = t * 16 + mrow;
        const float bv = bias[col];
        #pragma unroll
        for (int r = 0; r < 8; ++r) {
            const int orow = tile * 16 + half * 8 + r;
            float v = acc[t][r] + bv;
            if (RELU) v = v > 0.0f ? v : 0.0f;
            Out[(long)orow * DIM_H + col] = v;
        }
    }
}

// ---------------------------------------------------------------------------
// Layer 3 (128 -> 1): out[n] = deg_inv[n]*sum(agg[n]*W3l) + sum(h[n]*W3r) + b3.
// Wave-per-node dot product with wave32 shuffle reduction.
// ---------------------------------------------------------------------------
__global__ __launch_bounds__(256)
void sage_out_kernel(const float* __restrict__ agg, const float* __restrict__ h,
                     const float* __restrict__ deg_inv,
                     const float* __restrict__ W3l, const float* __restrict__ W3r,
                     const float* __restrict__ b3,
                     float* __restrict__ out, int nN) {
    const int wave = threadIdx.x >> 5;
    const int lane = threadIdx.x & 31;
    const int n = blockIdx.x * 8 + wave;
    if (n >= nN) return;                    // wave-uniform, no barriers below
    const float di = deg_inv[n];
    float s = 0.0f;
    #pragma unroll
    for (int d = lane; d < DIM_H; d += 32)
        s += agg[(long)n * DIM_H + d] * di * W3l[d] + h[(long)n * DIM_H + d] * W3r[d];
    #pragma unroll
    for (int off = 16; off > 0; off >>= 1) s += __shfl_down(s, off, 32);
    if (lane == 0) out[n] = s + b3[0];
}

// ---------------------------------------------------------------------------
extern "C" void kernel_launch(void* const* d_in, const int* in_sizes, int n_in,
                              void* d_out, int out_size, void* d_ws, size_t ws_size,
                              hipStream_t stream) {
    const float* x   = (const float*)d_in[0];
    const int*   ei  = (const int*)d_in[1];     // [2, N_EDGES]: row0=src, row1=dst
    const float* W1l = (const float*)d_in[2];
    const float* W1r = (const float*)d_in[3];
    const float* b1  = (const float*)d_in[4];
    const float* W2l = (const float*)d_in[5];
    const float* W2r = (const float*)d_in[6];
    const float* b2  = (const float*)d_in[7];
    const float* W3l = (const float*)d_in[8];
    const float* W3r = (const float*)d_in[9];
    const float* b3  = (const float*)d_in[10];
    float* out = (float*)d_out;

    const int N = N_NODES, E = N_EDGES;

    // Workspace (floats): deg_inv | agg[N x 128] | h[N x 128] | WT1 | WT2 (~103 MB)
    float*  ws  = (float*)d_ws;
    float*  deg = ws;                        // N floats (counts, then 1/max(deg,1))
    float*  agg = ws + (1 << 17);            // 128K-float pad for alignment
    float*  h   = agg + (long)N * DIM_H;
    __bf16* wt1 = (__bf16*)(h + (long)N * DIM_H);       // 2*128*32  bf16 (16 KB)
    __bf16* wt2 = wt1 + 2 * DIM_H * 32;                 // 2*128*128 bf16 (64 KB)

    const long aggN = (long)N * DIM_H;
    const int fillAggBlocks = (int)((aggN + 255) / 256);
    const int edgeBlocks    = (E + 255) / 256;
    const int gemmBlocks    = ((N / 16) + 7) / 8;        // 6250 tiles / 8 waves
    const int scatterBlocks = (int)(((long)E * 32 + 255) / 256);

    // One-shot weight packing (bf16, transposed, K-padded)
    pack_weights_kernel<DIM_IN><<<(2 * DIM_H * 32 + 255) / 256, 256, 0, stream>>>(W1l, W1r, wt1);
    pack_weights_kernel<DIM_H><<<(2 * DIM_H * DIM_H + 255) / 256, 256, 0, stream>>>(W2l, W2r, wt2);

    // deg_inv
    fill_kernel<<<(N + 255) / 256, 256, 0, stream>>>(deg, N, 0.0f);
    deg_count_kernel<<<edgeBlocks, 256, 0, stream>>>(ei, E, deg);
    deg_inv_kernel<<<(N + 255) / 256, 256, 0, stream>>>(deg, N);

    // Layer 1: agg = scatter(x); h = relu(mean(agg)@W1l + x@W1r + b1)
    fill_kernel<<<fillAggBlocks, 256, 0, stream>>>(agg, aggN, 0.0f);
    scatter15_kernel<<<edgeBlocks, 256, 0, stream>>>(ei, E, x, agg);
    sage_gemm_kernel<DIM_IN, true><<<gemmBlocks, 256, 0, stream>>>(
        agg, DIM_H, x, DIM_IN, wt1, b1, deg, h, N);

    // Layer 2: agg = scatter(h); h = relu(mean(agg)@W2l + h@W2r + b2)  (h in-place
    // safe: each row is fully read in the K loop before its strip is stored,
    // and exactly one wave owns each row)
    fill_kernel<<<fillAggBlocks, 256, 0, stream>>>(agg, aggN, 0.0f);
    scatter128_kernel<<<scatterBlocks, 256, 0, stream>>>(ei, E, h, agg);
    sage_gemm_kernel<DIM_H, true><<<gemmBlocks, 256, 0, stream>>>(
        agg, DIM_H, h, DIM_H, wt2, b2, deg, h, N);

    // Layer 3: agg = scatter(h); out = mean(agg)@W3l + h@W3r + b3
    fill_kernel<<<fillAggBlocks, 256, 0, stream>>>(agg, aggN, 0.0f);
    scatter128_kernel<<<scatterBlocks, 256, 0, stream>>>(ei, E, h, agg);
    sage_out_kernel<<<(N + 7) / 8, 256, 0, stream>>>(agg, h, deg, W3l, W3r, b3, out, N);
}
